// NeRF_12567074308167
// MI455X (gfx1250) — matrix operand order repaired and checked by reading, not compiled, not run on hardware
//
#include <hip/hip_runtime.h>
#include <hip/hip_bf16.h>
#include <math.h>

typedef _Float16 half_t;
typedef __attribute__((ext_vector_type(8)))  _Float16 v8h;
typedef __attribute__((ext_vector_type(16))) _Float16 v16h;
typedef __attribute__((ext_vector_type(8)))  float    v8f;
// under-aligned float2 for pairwise gather taps (4B-aligned b64 loads)
typedef float f2u __attribute__((ext_vector_type(2))) __attribute__((aligned(4)));

#define PTS_PER_BLK 128
#define THREADS     256
#define K1          96      // padded layer-1 K (87 real)
#define FS          136     // feat / h row stride (halves): 272B, 16B-aligned, 4-bank stride
#define W1S         104     // W1' row stride (halves): 208B, 16B-aligned
#define W2S         136
#define W3S         136

// ws layout (halves)
#define WS_W1_OFF   0
#define WS_W1_H     (128 * W1S)           // 13312
#define WS_W2_OFF   WS_W1_H
#define WS_W2_H     (128 * W2S)           // 17408
#define WS_W3_OFF   (WS_W2_OFF + WS_W2_H) // 30720
#define WS_W3_H     (16 * W3S)            // 2176
#define WS_TOTAL_H  (WS_W3_OFF + WS_W3_H) // 32896 halves = 65792 B

// LDS layout (bytes)
#define SM_W1_OFF   0
#define SM_W2_OFF   (SM_W1_OFF + WS_W1_H * 2)   // 26624
#define SM_W3_OFF   (SM_W2_OFF + WS_W2_H * 2)   // 61440
#define SM_FEAT_OFF (SM_W3_OFF + WS_W3_H * 2)   // 65792
#define SM_H1_OFF   (SM_FEAT_OFF + PTS_PER_BLK * FS * 2) // 100608
#define SM_TOTAL    (SM_H1_OFF + PTS_PER_BLK * FS * 2)   // 135424

// ---------------------------------------------------------------------------
// Prep kernel: build fused f16 weights in workspace.
//   W1'[n][k] : k<48   -> sum_j W1[n][j]*basis_W[j][k]   (basis folded in)
//               k<51   -> W1[n][27 + (k-48)]             (ray dims)
//               k<87   -> W1[n][30 + (k-51)]             (positional enc)
//               else 0
// ---------------------------------------------------------------------------
__global__ void nerf_prep_weights(const float* __restrict__ W1,
                                  const float* __restrict__ basisW,
                                  const float* __restrict__ W2,
                                  const float* __restrict__ W3,
                                  half_t* __restrict__ ws) {
  const int tid = threadIdx.x;
  // W1 fused (128 x 96)
  for (int idx = tid; idx < 128 * K1; idx += THREADS) {
    const int n = idx / K1;
    const int k = idx % K1;
    float v;
    if (k < 48) {
      const float* w1r = W1 + n * 66;
      float acc = 0.f;
#pragma unroll
      for (int j = 0; j < 27; ++j) acc += w1r[j] * basisW[j * 48 + k];
      v = acc;
    } else if (k < 51) {
      v = W1[n * 66 + 27 + (k - 48)];
    } else if (k < 87) {
      v = W1[n * 66 + 30 + (k - 51)];
    } else {
      v = 0.f;
    }
    ws[WS_W1_OFF + n * W1S + k] = (half_t)v;
  }
  // W2 (128 x 128)
  for (int idx = tid; idx < 128 * 128; idx += THREADS) {
    const int n = idx >> 7, k = idx & 127;
    ws[WS_W2_OFF + n * W2S + k] = (half_t)W2[n * 128 + k];
  }
  // W3 (3 x 128, zero-padded to 16 rows)
  for (int idx = tid; idx < 16 * 128; idx += THREADS) {
    const int n = idx >> 7, k = idx & 127;
    ws[WS_W3_OFF + n * W3S + k] = (half_t)((n < 3) ? W3[n * 128 + k] : 0.f);
  }
}

// ---------------------------------------------------------------------------
// WMMA fragment loaders (CDNA5 16x16x32 f16 layouts, wave32).
// A 16x32: lanes 0-15 hold M=lane, halves = K[0..7]+K[16..23];
//          lanes 16-31 hold M=lane-16, halves = K[8..15]+K[24..31].
// B 32x16: lanes 0-15 hold N=lane, halves = K[0..7]+K[16..23];
//          lanes 16-31 hold N=lane-16, halves = K[8..15]+K[24..31].
// ---------------------------------------------------------------------------
__device__ __forceinline__ v16h cat8(v8h lo, v8h hi) {
  return __builtin_shufflevector(lo, hi, 0, 1, 2, 3, 4, 5, 6, 7,
                                 8, 9, 10, 11, 12, 13, 14, 15);
}

__device__ __forceinline__ v16h load_fragA(const half_t* buf, int stride,
                                           int row0, int kbase, int lane) {
  const int m = lane & 15;
  const int koff = (lane < 16) ? 0 : 8;
  const half_t* p = buf + (row0 + m) * stride + kbase + koff;
  v8h lo = *(const v8h*)p;
  v8h hi = *(const v8h*)(p + 16);
  return cat8(lo, hi);
}

__device__ __forceinline__ v16h load_fragB(const half_t* w, int stride,
                                           int n0, int kbase, int lane) {
  const int n = lane & 15;
  const int koff = (lane < 16) ? 0 : 8;
  const half_t* p = w + (n0 + n) * stride + kbase + koff;
  v8h lo = *(const v8h*)p;
  v8h hi = *(const v8h*)(p + 16);
  return cat8(lo, hi);
}

__device__ __forceinline__ v8f wmma(v16h a, v16h b, v8f c) {
  return __builtin_amdgcn_wmma_f32_16x16x32_f16(false, a, false, b,
                                                (short)0, c, false, false);
}

// ---------------------------------------------------------------------------
// Fused kernel: sampling + encoding -> f16 features in LDS -> 3-layer WMMA MLP
// ---------------------------------------------------------------------------
__global__ void __launch_bounds__(THREADS)
nerf_fused(const float* __restrict__ pts, const float* __restrict__ ray,
           const float* __restrict__ dpl, const float* __restrict__ dln,
           const float* __restrict__ cpl, const float* __restrict__ cln,
           const float* __restrict__ b1, const float* __restrict__ b2,
           const float* __restrict__ b3, const half_t* __restrict__ ws,
           float* __restrict__ out, int N) {
  extern __shared__ char smem[];
  half_t* w1L   = (half_t*)(smem + SM_W1_OFF);
  half_t* w2L   = (half_t*)(smem + SM_W2_OFF);
  half_t* w3L   = (half_t*)(smem + SM_W3_OFF);
  half_t* featL = (half_t*)(smem + SM_FEAT_OFF);  // [128][FS]; reused as h2
  half_t* h1L   = (half_t*)(smem + SM_H1_OFF);    // [128][FS]

  const int tid = threadIdx.x;

  // ---- stage weights ws -> LDS (65792 B = 4112 x 16B) ----
  {
    const uint4* src = (const uint4*)ws;
    uint4* dst = (uint4*)smem;
    for (int i = tid; i < WS_TOTAL_H / 8; i += THREADS) dst[i] = src[i];
  }

  // ---- sampling + feature build: 2 threads / point, 8 channels each ----
  // Bilinear x-pairs and line pairs fetched as 4B-aligned float2 (b64 loads):
  // pair base clamped to 126; at the ix0==127 edge wx==0 so the v01 term is
  // weight-zero and the select keeps v00 exact.
  {
    const int p = tid >> 1;
    const int hs = tid & 1;             // channel half selector
    const int gp0 = blockIdx.x * PTS_PER_BLK + p;
    const int gp = (gp0 < N) ? gp0 : (N - 1);
    const float x = pts[gp * 3 + 0];
    const float y = pts[gp * 3 + 1];
    const float z = pts[gp * 3 + 2];
    // MAT_MODE = ((0,1),(0,2),(1,2)); VEC_MODE = (2,1,0)
    const float gax[3] = {x, x, y};
    const float gay[3] = {y, z, z};
    const float glv[3] = {z, y, x};

    float sig = 0.f;
#pragma unroll
    for (int i = 0; i < 3; ++i) {
      float fx = (gax[i] + 1.f) * 63.5f;
      float fy = (gay[i] + 1.f) * 63.5f;
      float fl = (glv[i] + 1.f) * 63.5f;
      int ix0 = (int)floorf(fx); ix0 = min(max(ix0, 0), 127);
      int iy0 = (int)floorf(fy); iy0 = min(max(iy0, 0), 127);
      int il0 = (int)floorf(fl); il0 = min(max(il0, 0), 127);
      const int iy1 = min(iy0 + 1, 127);
      const float wx = fx - (float)ix0;
      const float wy = fy - (float)iy0;
      const float wl = fl - (float)il0;
      const float w00 = (1.f - wx) * (1.f - wy), w01 = wx * (1.f - wy);
      const float w10 = (1.f - wx) * wy,         w11 = wx * wy;
      const float wl0 = 1.f - wl;
      const int bx = min(ix0, 126);
      const int bl = min(il0, 126);
      const bool selx = (ix0 != bx);    // true only at the 127 edge
      const bool sell = (il0 != bl);
      const int r0 = iy0 * 128 + bx;
      const int r1 = iy1 * 128 + bx;
#pragma unroll
      for (int c8 = 0; c8 < 8; ++c8) {
        const int cc = hs * 8 + c8;
        const int ch = i * 16 + cc;
        // density
        {
          const float* pl = dpl + ch * 16384;
          const f2u t = *(const f2u*)(pl + r0);
          const f2u bm = *(const f2u*)(pl + r1);
          const float v00 = selx ? t.y : t.x;
          const float v10 = selx ? bm.y : bm.x;
          const float dv = v00 * w00 + t.y * w01 + v10 * w10 + bm.y * w11;
          const f2u lp = *(const f2u*)(dln + ch * 128 + bl);
          const float l0 = sell ? lp.y : lp.x;
          sig += dv * (l0 * wl0 + lp.y * wl);
        }
        // color: s = plane * line  -> feature k = ch
        {
          const float* pl = cpl + ch * 16384;
          const f2u t = *(const f2u*)(pl + r0);
          const f2u bm = *(const f2u*)(pl + r1);
          const float v00 = selx ? t.y : t.x;
          const float v10 = selx ? bm.y : bm.x;
          const float cv = v00 * w00 + t.y * w01 + v10 * w10 + bm.y * w11;
          const f2u lp = *(const f2u*)(cln + ch * 128 + bl);
          const float l0 = sell ? lp.y : lp.x;
          featL[p * FS + ch] = (half_t)(cv * (l0 * wl0 + lp.y * wl));
        }
      }
    }
    sig += __shfl_xor(sig, 1, 32);
    if (hs == 0) {
      if (gp0 < N) out[3 * N + gp0] = sig;
      // ray + positional encoding, k = 48..86; zero pad 87..95
      const float rx = ray[gp * 3 + 0];
      const float ry = ray[gp * 3 + 1];
      const float rz = ray[gp * 3 + 2];
      half_t* f = featL + p * FS;
      f[48] = (half_t)rx; f[49] = (half_t)ry; f[50] = (half_t)rz;
      const float rv[3] = {rx, ry, rz};
#pragma unroll
      for (int d = 0; d < 3; ++d) {
        float fr = 3.14159265358979f;
#pragma unroll
        for (int j = 0; j < 6; ++j) {
          const float sp = rv[d] * fr;
          f[51 + d * 12 + j]     = (half_t)sinf(sp);
          f[51 + d * 12 + 6 + j] = (half_t)cosf(sp);
          fr *= 2.f;
        }
      }
#pragma unroll
      for (int k = 87; k < K1; ++k) f[k] = (half_t)0.f;
    }
  }
  __syncthreads();

  const int wv = tid >> 5;     // wave id: owns M-tile rows [16*wv, 16*wv+16)
  const int lane = tid & 31;
  const int n = lane & 15;
  const int mb = (lane < 16) ? 0 : 8;

  // ---- layer 1: [128 x 96] @ [96 x 128] -> ReLU -> h1 ----
  {
    v8f acc[8];
#pragma unroll
    for (int t = 0; t < 8; ++t) acc[t] = (v8f)0.f;
#pragma unroll
    for (int kt = 0; kt < 3; ++kt) {
      const v16h a = load_fragA(featL, FS, wv * 16, kt * 32, lane);
#pragma unroll
      for (int nt = 0; nt < 8; ++nt) {
        const v16h b = load_fragB(w1L, W1S, nt * 16, kt * 32, lane);
        acc[nt] = wmma(a, b, acc[nt]);
      }
    }
#pragma unroll
    for (int nt = 0; nt < 8; ++nt) {
      const float bias = b1[nt * 16 + n];
#pragma unroll
      for (int r = 0; r < 8; ++r) {
        const float hv = fmaxf(acc[nt][r] + bias, 0.f);
        h1L[(wv * 16 + mb + r) * FS + nt * 16 + n] = (half_t)hv;
      }
    }
  }
  __syncthreads();

  // ---- layer 2: [128 x 128] @ [128 x 128] -> ReLU -> h2 (reuse featL) ----
  {
    v8f acc[8];
#pragma unroll
    for (int t = 0; t < 8; ++t) acc[t] = (v8f)0.f;
#pragma unroll
    for (int kt = 0; kt < 4; ++kt) {
      const v16h a = load_fragA(h1L, FS, wv * 16, kt * 32, lane);
#pragma unroll
      for (int nt = 0; nt < 8; ++nt) {
        const v16h b = load_fragB(w2L, W2S, nt * 16, kt * 32, lane);
        acc[nt] = wmma(a, b, acc[nt]);
      }
    }
#pragma unroll
    for (int nt = 0; nt < 8; ++nt) {
      const float bias = b2[nt * 16 + n];
#pragma unroll
      for (int r = 0; r < 8; ++r) {
        const float hv = fmaxf(acc[nt][r] + bias, 0.f);
        featL[(wv * 16 + mb + r) * FS + nt * 16 + n] = (half_t)hv;
      }
    }
  }
  __syncthreads();

  // ---- layer 3: [128 x 128] @ [128 x 16(pad)] -> sigmoid -> rgb ----
  {
    v8f acc = (v8f)0.f;
#pragma unroll
    for (int kt = 0; kt < 4; ++kt) {
      const v16h a = load_fragA(featL, FS, wv * 16, kt * 32, lane);
      const v16h b = load_fragB(w3L, W3S, 0, kt * 32, lane);
      acc = wmma(a, b, acc);
    }
    if (n < 3) {
      const float bias = b3[n];
#pragma unroll
      for (int r = 0; r < 8; ++r) {
        const int gp = blockIdx.x * PTS_PER_BLK + wv * 16 + mb + r;
        if (gp < N) {
          const float v = acc[r] + bias;
          out[gp * 3 + n] = 1.f / (1.f + expf(-v));
        }
      }
    }
  }
}

extern "C" void kernel_launch(void* const* d_in, const int* in_sizes, int n_in,
                              void* d_out, int out_size, void* d_ws, size_t ws_size,
                              hipStream_t stream) {
  const float* pts    = (const float*)d_in[0];
  const float* ray    = (const float*)d_in[1];
  const float* dpl    = (const float*)d_in[2];
  const float* dln    = (const float*)d_in[3];
  const float* cpl    = (const float*)d_in[4];
  const float* cln    = (const float*)d_in[5];
  const float* basisW = (const float*)d_in[6];
  const float* W1     = (const float*)d_in[7];
  const float* b1     = (const float*)d_in[8];
  const float* W2     = (const float*)d_in[9];
  const float* b2     = (const float*)d_in[10];
  const float* W3     = (const float*)d_in[11];
  const float* b3     = (const float*)d_in[12];
  float* out = (float*)d_out;
  half_t* ws = (half_t*)d_ws;

  const int N = in_sizes[0] / 3;

  nerf_prep_weights<<<1, THREADS, 0, stream>>>(W1, basisW, W2, W3, ws);

  const int blocks = (N + PTS_PER_BLK - 1) / PTS_PER_BLK;
  nerf_fused<<<blocks, THREADS, SM_TOTAL, stream>>>(
      pts, ray, dpl, dln, cpl, cln, b1, b2, b3, ws, out, N);
}
